// MoEBlock_88038239633778
// MI455X (gfx1250) — matrix operand order repaired and checked by reading, not compiled, not run on hardware
//
#include <hip/hip_runtime.h>

#define T_TOK 4096
#define HID   1024
#define INTER 1408
#define NEXP  8

typedef __attribute__((ext_vector_type(16))) __bf16 v16bf;
typedef __attribute__((ext_vector_type(8)))  __bf16 v8bf;
typedef __attribute__((ext_vector_type(4)))  __bf16 v4bf;
typedef __attribute__((ext_vector_type(8)))  float  v8f;
typedef __attribute__((ext_vector_type(4)))  float  v4f;

union ABu { v16bf v; v8bf h[2]; __bf16 e[16]; };

__device__ __forceinline__ float fast_silu(float g) {
    // g * sigmoid(g); v_rcp_f32 instead of IEEE-div Newton chain
    return g * __builtin_amdgcn_rcpf(1.f + __expf(-g));
}

// ---------------- zero output segment ----------------
__global__ __launch_bounds__(256) void k_zero_out(float* out) {
    size_t idx = (size_t)blockIdx.x * 1024 + threadIdx.x * 4;
    v4f z = {};
    *(v4f*)(out + idx) = z;
}

// ---------------- fp32 -> bf16 bulk convert (x and optionally weights) ----------------
__global__ __launch_bounds__(256) void k_cvt(const float* __restrict__ src, __bf16* __restrict__ dst) {
    size_t idx = (size_t)blockIdx.x * 1024 + threadIdx.x * 4;
    v4f xv = *(const v4f*)(src + idx);
    v4bf o;
    o[0] = (__bf16)xv[0]; o[1] = (__bf16)xv[1];
    o[2] = (__bf16)xv[2]; o[3] = (__bf16)xv[3];
    *(v4bf*)(dst + idx) = o;
}

// ---------------- router: logits + softmax top-2 ----------------
__global__ __launch_bounds__(256) void k_router(const float* __restrict__ x,
                                                const float* __restrict__ gw,
                                                float* __restrict__ logits,
                                                int* __restrict__ sel,
                                                float* __restrict__ selw) {
    int t = blockIdx.x * 256 + threadIdx.x;
    const float* xr = x + (size_t)t * HID;
    float acc[NEXP];
#pragma unroll
    for (int e = 0; e < NEXP; ++e) acc[e] = 0.f;
    for (int k = 0; k < HID; k += 4) {
        v4f xv = *(const v4f*)(xr + k);
#pragma unroll
        for (int e = 0; e < NEXP; ++e) {
            v4f wv = *(const v4f*)(gw + (size_t)e * HID + k);
            acc[e] = fmaf(xv[0], wv[0], acc[e]);
            acc[e] = fmaf(xv[1], wv[1], acc[e]);
            acc[e] = fmaf(xv[2], wv[2], acc[e]);
            acc[e] = fmaf(xv[3], wv[3], acc[e]);
        }
    }
#pragma unroll
    for (int e = 0; e < NEXP; ++e) logits[(size_t)t * NEXP + e] = acc[e];
    float m = acc[0];
#pragma unroll
    for (int e = 1; e < NEXP; ++e) m = fmaxf(m, acc[e]);
    float p[NEXP];
#pragma unroll
    for (int e = 0; e < NEXP; ++e) p[e] = __expf(acc[e] - m);
    int i1 = 0; float b1 = p[0];
#pragma unroll
    for (int e = 1; e < NEXP; ++e) if (p[e] > b1) { b1 = p[e]; i1 = e; }
    int i2 = -1; float b2 = -1.f;
#pragma unroll
    for (int e = 0; e < NEXP; ++e) if (e != i1 && p[e] > b2) { b2 = p[e]; i2 = e; }
    float s = b1 + b2;
    sel[2 * t]      = i1;     sel[2 * t + 1]  = i2;
    selw[2 * t]     = b1 / s; selw[2 * t + 1] = b2 / s;
}

// ---------------- per-expert gather lists (deterministic block scan) ----------------
__global__ __launch_bounds__(256) void k_build_lists(const int* __restrict__ sel,
                                                     const float* __restrict__ selw,
                                                     int* __restrict__ list,
                                                     float* __restrict__ wgt,
                                                     int* __restrict__ counts) {
    int e = blockIdx.x;
    int tid = threadIdx.x;
    __shared__ int s[256];
    __shared__ int baseSh;
    if (tid == 0) baseSh = 0;
    __syncthreads();
    for (int c = 0; c < T_TOK / 256; ++c) {
        int t = c * 256 + tid;
        int f = 0; float w = 0.f;
        int s0 = sel[2 * t], s1 = sel[2 * t + 1];
        if (s0 == e)      { f = 1; w = selw[2 * t]; }
        else if (s1 == e) { f = 1; w = selw[2 * t + 1]; }
        s[tid] = f;
        __syncthreads();
        for (int off = 1; off < 256; off <<= 1) {
            int v = (tid >= off) ? s[tid - off] : 0;
            __syncthreads();
            s[tid] += v;
            __syncthreads();
        }
        int incl  = s[tid];
        int total = s[255];
        int base  = baseSh;
        if (f) {
            int pos = base + incl - 1;
            list[e * T_TOK + pos] = t;
            wgt [e * T_TOK + pos] = w;
        }
        __syncthreads();
        if (tid == 0) baseSh = base + total;
        __syncthreads();
    }
    int count  = baseSh;
    int padded = (count + 63) & ~63;
    for (int i = count + tid; i < padded; i += 256) {
        list[e * T_TOK + i] = -1;
        wgt [e * T_TOK + i] = 0.f;
    }
    if (tid == 0) { counts[e] = count; counts[8 + e] = padded; }
}

__global__ void k_hbase(int* counts) {
    if (threadIdx.x == 0 && blockIdx.x == 0) {
        int b = 0;
        for (int e = 0; e < NEXP; ++e) { counts[16 + e] = b; b += counts[8 + e]; }
    }
}

// ---------------- B-operand loader: 16 contiguous k-values -> v16bf ----------------
__device__ __forceinline__ void load_b16(const float* p, ABu& B) {
#pragma unroll
    for (int i = 0; i < 4; ++i) {
        v4f f = *(const v4f*)(p + i * 4);
#pragma unroll
        for (int j = 0; j < 4; ++j) B.e[i * 4 + j] = (__bf16)f[j];
    }
}
__device__ __forceinline__ void load_b16(const __bf16* p, ABu& B) {
    B.h[0] = *(const v8bf*)(p);
    B.h[1] = *(const v8bf*)(p + 8);
}

// ---------------- GEMM1: h = silu(x@Wg^T) * (x@Wu^T), grouped by expert ----------------
template <typename BT>
__global__ __launch_bounds__(256) void k_gemm1(const __bf16* __restrict__ xb,
                                               const BT* __restrict__ gate_proj,
                                               const BT* __restrict__ up_proj,
                                               const int* __restrict__ list,
                                               const int* __restrict__ counts,
                                               __bf16* __restrict__ hbuf) {
    const int e      = blockIdx.z;
    const int padded = counts[8 + e];
    const int mBase  = blockIdx.y * 64;
    if (mBase >= padded) return;
    const int nBase = blockIdx.x * 64;
    const int tid = threadIdx.x, lane = tid & 31, wv = tid >> 5;
    const int wm = wv >> 2, wn = wv & 3;
    const int hb = counts[16 + e];
    const int l15 = lane & 15;
    const bool hi = lane >= 16;

    int row0 = mBase + wm * 32 + l15;
    int tok0 = list[e * T_TOK + row0];      if (tok0 < 0) tok0 = 0;
    int tok1 = list[e * T_TOK + row0 + 16]; if (tok1 < 0) tok1 = 0;
    const __bf16* a0p = xb + (size_t)tok0 * HID;
    const __bf16* a1p = xb + (size_t)tok1 * HID;
    const int selA = hi ? 16 : 0;
    const int n = nBase + wn * 16 + l15;
    const BT* bgp = gate_proj + ((size_t)e * INTER + n) * HID;
    const BT* bup = up_proj   + ((size_t)e * INTER + n) * HID;
    const int selB = hi ? 16 : 0;

    v8f accg0 = {}, accu0 = {}, accg1 = {}, accu1 = {};
    for (int k0 = 0; k0 < HID; k0 += 32) {
        ABu A0, A1, Bg, Bu;
        A0.h[0] = *(const v8bf*)(a0p + k0 + selA);
        A0.h[1] = *(const v8bf*)(a0p + k0 + 8 + selA);
        A1.h[0] = *(const v8bf*)(a1p + k0 + selA);
        A1.h[1] = *(const v8bf*)(a1p + k0 + 8 + selA);
        load_b16(bgp + k0 + selB, Bg);
        load_b16(bup + k0 + selB, Bu);
        accg0 = __builtin_amdgcn_wmma_f32_16x16x32_bf16(false, A0.v, false, Bg.v, (short)0, accg0, false, false);
        accu0 = __builtin_amdgcn_wmma_f32_16x16x32_bf16(false, A0.v, false, Bu.v, (short)0, accu0, false, false);
        accg1 = __builtin_amdgcn_wmma_f32_16x16x32_bf16(false, A1.v, false, Bg.v, (short)0, accg1, false, false);
        accu1 = __builtin_amdgcn_wmma_f32_16x16x32_bf16(false, A1.v, false, Bu.v, (short)0, accu1, false, false);
    }

    const int outN = nBase + wn * 16 + l15;
    const int mrow = hi ? 8 : 0;
#pragma unroll
    for (int r = 0; r < 8; ++r) {
        int m0 = mBase + wm * 32 + mrow + r;
        hbuf[(size_t)(hb + m0) * INTER + outN] = (__bf16)(fast_silu(accg0[r]) * accu0[r]);
        int m1 = m0 + 16;
        hbuf[(size_t)(hb + m1) * INTER + outN] = (__bf16)(fast_silu(accg1[r]) * accu1[r]);
    }
}

// ---------------- GEMM2: out[tok] += w * (h @ Wd^T), scatter by token ----------------
template <typename BT>
__global__ __launch_bounds__(256) void k_gemm2(const __bf16* __restrict__ hbuf,
                                               const BT* __restrict__ down_proj,
                                               const int* __restrict__ list,
                                               const float* __restrict__ wgt,
                                               const int* __restrict__ counts,
                                               float* __restrict__ out) {
    const int e      = blockIdx.z;
    const int padded = counts[8 + e];
    const int mBase  = blockIdx.y * 64;
    if (mBase >= padded) return;
    const int nBase = blockIdx.x * 64;
    const int tid = threadIdx.x, lane = tid & 31, wv = tid >> 5;
    const int wm = wv >> 2, wn = wv & 3;
    const int hb = counts[16 + e];
    const int l15 = lane & 15;
    const bool hi = lane >= 16;

    int row0 = mBase + wm * 32 + l15;
    const __bf16* a0p = hbuf + (size_t)(hb + row0) * INTER;
    const __bf16* a1p = hbuf + (size_t)(hb + row0 + 16) * INTER;
    const int selA = hi ? 16 : 0;
    const int n = nBase + wn * 16 + l15;
    const BT* bdp = down_proj + ((size_t)e * HID + n) * INTER;
    const int selB = hi ? 16 : 0;

    v8f acc0 = {}, acc1 = {};
    for (int k0 = 0; k0 < INTER; k0 += 32) {
        ABu A0, A1, Bd;
        A0.h[0] = *(const v8bf*)(a0p + k0 + selA);
        A0.h[1] = *(const v8bf*)(a0p + k0 + 8 + selA);
        A1.h[0] = *(const v8bf*)(a1p + k0 + selA);
        A1.h[1] = *(const v8bf*)(a1p + k0 + 8 + selA);
        load_b16(bdp + k0 + selB, Bd);
        acc0 = __builtin_amdgcn_wmma_f32_16x16x32_bf16(false, A0.v, false, Bd.v, (short)0, acc0, false, false);
        acc1 = __builtin_amdgcn_wmma_f32_16x16x32_bf16(false, A1.v, false, Bd.v, (short)0, acc1, false, false);
    }

    const int col = nBase + wn * 16 + l15;
    const int mrow = hi ? 8 : 0;
#pragma unroll
    for (int r = 0; r < 8; ++r) {
        int row = mBase + wm * 32 + mrow + r;
        int tok = list[e * T_TOK + row];
        float w = wgt[e * T_TOK + row];
        if (tok >= 0) atomicAdd(out + (size_t)tok * HID + col, w * acc0[r]);
        row += 16;
        tok = list[e * T_TOK + row];
        w   = wgt[e * T_TOK + row];
        if (tok >= 0) atomicAdd(out + (size_t)tok * HID + col, w * acc1[r]);
    }
}

// ---------------- host launcher ----------------
extern "C" void kernel_launch(void* const* d_in, const int* in_sizes, int n_in,
                              void* d_out, int out_size, void* d_ws, size_t ws_size,
                              hipStream_t stream) {
    const float* x  = (const float*)d_in[0];
    const float* gw = (const float*)d_in[1];
    const float* gp = (const float*)d_in[2];
    const float* up = (const float*)d_in[3];
    const float* dp = (const float*)d_in[4];
    float* out = (float*)d_out;

    const size_t W_ELEMS = (size_t)NEXP * INTER * HID;   // 11,534,336 per matrix

    char* ws = (char*)d_ws;
    size_t off = 0;
    __bf16* xb   = (__bf16*)(ws + off); off += (size_t)T_TOK * HID * 2;   // 8 MB
    int*    sel  = (int*)   (ws + off); off += (size_t)T_TOK * 2 * 4;
    float*  selw = (float*) (ws + off); off += (size_t)T_TOK * 2 * 4;
    int*    list = (int*)   (ws + off); off += (size_t)NEXP * T_TOK * 4;
    float*  wgt  = (float*) (ws + off); off += (size_t)NEXP * T_TOK * 4;
    int*    cnts = (int*)   (ws + off); off += 256;
    __bf16* hbuf = (__bf16*)(ws + off); off += (size_t)8704 * INTER * 2;  // ~24.5 MB
    __bf16* gpb  = (__bf16*)(ws + off); off += W_ELEMS * 2;               // 23 MB each
    __bf16* upb  = (__bf16*)(ws + off); off += W_ELEMS * 2;
    __bf16* dpb  = (__bf16*)(ws + off); off += W_ELEMS * 2;
    const bool bf16_weights = (ws_size >= off);

    float* logits = out + (size_t)T_TOK * HID;

    k_zero_out   <<<4096, 256, 0, stream>>>(out);
    k_cvt        <<<4096, 256, 0, stream>>>(x, xb);
    k_router     <<<T_TOK / 256, 256, 0, stream>>>(x, gw, logits, sel, selw);
    k_build_lists<<<NEXP, 256, 0, stream>>>(sel, selw, list, wgt, cnts);
    k_hbase      <<<1, 32, 0, stream>>>(cnts);

    dim3 g1(INTER / 64, T_TOK / 64, NEXP);
    dim3 g2(HID / 64,   T_TOK / 64, NEXP);
    if (bf16_weights) {
        int wblocks = (int)(W_ELEMS / 1024);             // 11264
        k_cvt<<<wblocks, 256, 0, stream>>>(gp, gpb);
        k_cvt<<<wblocks, 256, 0, stream>>>(up, upb);
        k_cvt<<<wblocks, 256, 0, stream>>>(dp, dpb);
        k_gemm1<__bf16><<<g1, 256, 0, stream>>>(xb, gpb, upb, list, cnts, hbuf);
        k_gemm2<__bf16><<<g2, 256, 0, stream>>>(hbuf, dpb, list, wgt, cnts, out);
    } else {
        k_gemm1<float><<<g1, 256, 0, stream>>>(xb, gp, up, list, cnts, hbuf);
        k_gemm2<float><<<g2, 256, 0, stream>>>(hbuf, dp, list, wgt, cnts, out);
    }
}
